// GraphNetwork_8615704396469
// MI455X (gfx1250) — matrix operand order repaired and checked by reading, not compiled, not run on hardware
//
#include <hip/hip_runtime.h>
#include <hip/hip_bf16.h>

// ---------------------------------------------------------------------------
// Problem constants (match reference)
// ---------------------------------------------------------------------------
#define NN    50000
#define EE    800000
#define FEAT  256
#define HID   128
#define RR    16
#define NB    30
#define GG    2048
#define CC    10

typedef __attribute__((ext_vector_type(16))) __bf16 v16bf;
typedef __attribute__((ext_vector_type(8)))  float  v8f;

// ---------------------------------------------------------------------------
// WMMA fragment helpers (A/B in contiguous k = 16h+i order, C/D per ISA 7.12.2, wave32)
// ---------------------------------------------------------------------------
// A-matrix 16x32 bf16 from row-major f32 (on-the-fly convert).
// lane L: m = L&15, h = L>>4.  element i: K = h*16+i
__device__ inline v16bf load_a_f32(const float* __restrict__ A, int lda) {
  const int lane = threadIdx.x & 31;
  const int m = lane & 15, h = lane >> 4;
  v16bf a;
#pragma unroll
  for (int i = 0; i < 16; ++i) {
    const int v = i >> 1, p = i & 1;
    const int k = h * 16 + v * 2 + p;
    a[i] = (__bf16)A[(long long)m * lda + k];
  }
  return a;
}

// B-matrix 32x16 bf16 fragment from TRANSPOSED bf16 storage Bt[N][K]:
// lane L: n = L&15, h = L>>4; element i holds K = h*16+i  -> 32 contiguous bytes
// per lane => two global_load_b128 instead of 16 strided u16 loads.
__device__ inline v16bf load_b_bf16t(const __bf16* __restrict__ Bt, int ldt) {
  const int lane = threadIdx.x & 31;
  const int n = lane & 15, h = lane >> 4;
  const __bf16* p = Bt + (long long)n * ldt + h * 16;
  v16bf b;
#pragma unroll
  for (int i = 0; i < 16; ++i) b[i] = p[i];
  return b;
}

// C/D f32 16x16: lane L -> col = L&15; VGPR r -> row = r + 8*(L>>4)
#define CD_ROW(r)  ((r) + ((threadIdx.x & 31) >> 4) * 8)
#define CD_COL     ((threadIdx.x) & 15)

#define WMMA_BF16(a, b, c) \
  __builtin_amdgcn_wmma_f32_16x16x32_bf16(false, (a), false, (b), (short)0, (c), false, false)

// ---------------------------------------------------------------------------
// Utility kernels
// ---------------------------------------------------------------------------
__global__ void k_fill(float* p, long long n, float v) {
  long long i = (long long)blockIdx.x * blockDim.x + threadIdx.x;
  const long long st = (long long)gridDim.x * blockDim.x;
  for (; i < n; i += st) p[i] = v;
}

// f32 row-major [K][N]  ->  bf16 transposed [N][K]
__global__ void k_cvt_bf16_t(const float* __restrict__ s, __bf16* __restrict__ d,
                             int K, int N) {
  int i = blockIdx.x * blockDim.x + threadIdx.x;
  const int st = gridDim.x * blockDim.x;
  const int total = K * N;
  for (; i < total; i += st) {
    const int k = i / N, n = i - k * N;
    d[(long long)n * K + k] = (__bf16)s[i];
  }
}

// ---------------------------------------------------------------------------
// Stage 1: agg[t] += ew[e] * x[s]   (float4-vectorized edge scatter)
// ---------------------------------------------------------------------------
__global__ void k_edge_agg1(const float4* __restrict__ x4, const float* __restrict__ ew,
                            const int* __restrict__ src, const int* __restrict__ tgt,
                            float* __restrict__ agg) {
  const long long total = (long long)EE * (FEAT / 4);
  long long i = (long long)blockIdx.x * blockDim.x + threadIdx.x;
  const long long st = (long long)gridDim.x * blockDim.x;
  for (; i < total; i += st) {
    const long long e = i >> 6;          // /64 quads per edge
    const int q = (int)(i & 63);
    const int s = src[e], t = tgt[e];
    const float w = ew[e];
    const float4 v = x4[(long long)s * (FEAT / 4) + q];
    float* dst = &agg[(long long)t * FEAT + q * 4];
    atomicAdd(dst + 0, w * v.x);
    atomicAdd(dst + 1, w * v.y);
    atomicAdd(dst + 2, w * v.z);
    atomicAdd(dst + 3, w * v.w);
  }
}

// ---------------------------------------------------------------------------
// Stage 2: h1 = relu(agg@w_rel + x@w_root + b_rel)  [N,128]
// one wave per 16-row strip: A loaded once per K-step, 8 WMMAs reuse it
// ---------------------------------------------------------------------------
__global__ void k_h1(const float* __restrict__ agg, const float* __restrict__ x,
                     const __bf16* __restrict__ wrelT, const __bf16* __restrict__ wrootT,
                     const float* __restrict__ brel, float* __restrict__ h1) {
  const int wid = blockIdx.x * (blockDim.x >> 5) + (threadIdx.x >> 5);
  if (wid >= NN / 16) return;                      // wave-uniform
  const int row0 = wid * 16;
  v8f c[8];
#pragma unroll
  for (int ot = 0; ot < 8; ++ot) {
    const float bs = brel[ot * 16 + CD_COL];
#pragma unroll
    for (int i = 0; i < 8; ++i) c[ot][i] = bs;
  }
  for (int k0 = 0; k0 < FEAT; k0 += 32) {
    v16bf a = load_a_f32(agg + (long long)row0 * FEAT + k0, FEAT);
#pragma unroll
    for (int ot = 0; ot < 8; ++ot) {
      v16bf b = load_b_bf16t(wrelT + (long long)(ot * 16) * FEAT + k0, FEAT);
      c[ot] = WMMA_BF16(a, b, c[ot]);
    }
  }
  for (int k0 = 0; k0 < FEAT; k0 += 32) {
    v16bf a = load_a_f32(x + (long long)row0 * FEAT + k0, FEAT);
#pragma unroll
    for (int ot = 0; ot < 8; ++ot) {
      v16bf b = load_b_bf16t(wrootT + (long long)(ot * 16) * FEAT + k0, FEAT);
      c[ot] = WMMA_BF16(a, b, c[ot]);
    }
  }
#pragma unroll
  for (int ot = 0; ot < 8; ++ot)
#pragma unroll
    for (int i = 0; i < 8; ++i)
      h1[(long long)(row0 + CD_ROW(i)) * HID + ot * 16 + CD_COL] = fmaxf(c[ot][i], 0.f);
}

// ---------------------------------------------------------------------------
// Stage 3: w_r[r,k,o] = sum_b comp[r,b]*bases[b,k,o] -> bf16 TRANSPOSED [r][o][k]
// ---------------------------------------------------------------------------
__global__ void k_wr(const float* __restrict__ comp, const float* __restrict__ bases,
                     __bf16* __restrict__ wrT) {
  const int idx = blockIdx.x * blockDim.x + threadIdx.x;   // R*HID*HID = 262144
  if (idx >= RR * HID * HID) return;
  const int o = idx & 127, k = (idx >> 7) & 127, r = idx >> 14;
  float s = 0.f;
#pragma unroll
  for (int b = 0; b < NB; ++b)
    s += comp[r * NB + b] * bases[((long long)b * HID + k) * HID + o];
  wrT[((long long)r * HID + o) * HID + k] = (__bf16)s;
}

// ---------------------------------------------------------------------------
// Stage 4: xw = h1 @ w_r  -> [N, R*128]  (26-GFLOP GEMM)
// one wave per (16-row strip, relation): 8-tile A reuse, 32 WMMAs per wave
// ---------------------------------------------------------------------------
__global__ void k_xw(const float* __restrict__ h1, const __bf16* __restrict__ wrT,
                     float* __restrict__ xw) {
  const int wid = blockIdx.x * (blockDim.x >> 5) + (threadIdx.x >> 5);
  const int mt = wid >> 4;
  if (mt >= NN / 16) return;
  const int r = wid & 15;
  const int row0 = mt * 16;
  v8f c[8] = {};
  for (int k0 = 0; k0 < HID; k0 += 32) {
    v16bf a = load_a_f32(h1 + (long long)row0 * HID + k0, HID);
#pragma unroll
    for (int ot = 0; ot < 8; ++ot) {
      v16bf b = load_b_bf16t(wrT + ((long long)r * HID + ot * 16) * HID + k0, HID);
      c[ot] = WMMA_BF16(a, b, c[ot]);
    }
  }
#pragma unroll
  for (int ot = 0; ot < 8; ++ot)
#pragma unroll
    for (int i = 0; i < 8; ++i)
      xw[(long long)(row0 + CD_ROW(i)) * (RR * HID) + r * HID + ot * 16 + CD_COL] = c[ot][i];
}

// ---------------------------------------------------------------------------
// Stage 5: per-(target,relation) counts, then mean-normalized scatter (float4)
// ---------------------------------------------------------------------------
__global__ void k_edge_cnt(const int* __restrict__ tgt, const int* __restrict__ typ,
                           float* __restrict__ cnt) {
  int e = blockIdx.x * blockDim.x + threadIdx.x;
  const int st = gridDim.x * blockDim.x;
  for (; e < EE; e += st)
    atomicAdd(&cnt[(long long)tgt[e] * RR + typ[e]], 1.f);
}

__global__ void k_edge_agg2(const float4* __restrict__ xw4, const float* __restrict__ cnt,
                            const int* __restrict__ src, const int* __restrict__ tgt,
                            const int* __restrict__ typ, float* __restrict__ agg2) {
  const long long total = (long long)EE * (HID / 4);
  long long i = (long long)blockIdx.x * blockDim.x + threadIdx.x;
  const long long st = (long long)gridDim.x * blockDim.x;
  for (; i < total; i += st) {
    const long long e = i >> 5;          // /32 quads per edge
    const int q = (int)(i & 31);
    const int t = tgt[e], ty = typ[e];
    const float inv = 1.f / fmaxf(cnt[(long long)t * RR + ty], 1.f);
    const float4 v = xw4[(long long)src[e] * (RR * HID / 4) + ty * (HID / 4) + q];
    float* dst = &agg2[(long long)t * HID + q * 4];
    atomicAdd(dst + 0, inv * v.x);
    atomicAdd(dst + 1, inv * v.y);
    atomicAdd(dst + 2, inv * v.z);
    atomicAdd(dst + 3, inv * v.w);
  }
}

// ---------------------------------------------------------------------------
// Stage 6: h2 = relu(agg2 + h1@rgcn_root + rgcn_bias)  (WMMA, C preloaded)
// ---------------------------------------------------------------------------
__global__ void k_h2(const float* __restrict__ h1, const float* __restrict__ agg2,
                     const __bf16* __restrict__ wT, const float* __restrict__ bias,
                     float* __restrict__ h2) {
  const int wid = blockIdx.x * (blockDim.x >> 5) + (threadIdx.x >> 5);
  if (wid >= NN / 16) return;
  const int row0 = wid * 16;
  v8f c[8];
#pragma unroll
  for (int ot = 0; ot < 8; ++ot) {
    const float bs = bias[ot * 16 + CD_COL];
#pragma unroll
    for (int i = 0; i < 8; ++i)
      c[ot][i] = agg2[(long long)(row0 + CD_ROW(i)) * HID + ot * 16 + CD_COL] + bs;
  }
  for (int k0 = 0; k0 < HID; k0 += 32) {
    v16bf a = load_a_f32(h1 + (long long)row0 * HID + k0, HID);
#pragma unroll
    for (int ot = 0; ot < 8; ++ot) {
      v16bf b = load_b_bf16t(wT + (long long)(ot * 16) * HID + k0, HID);
      c[ot] = WMMA_BF16(a, b, c[ot]);
    }
  }
#pragma unroll
  for (int ot = 0; ot < 8; ++ot)
#pragma unroll
    for (int i = 0; i < 8; ++i)
      h2[(long long)(row0 + CD_ROW(i)) * HID + ot * 16 + CD_COL] = fmaxf(c[ot][i], 0.f);
}

// ---------------------------------------------------------------------------
// Stage 7: per-graph sum & max pooling of feat = [x | h2]  (384 dims, float4)
// ---------------------------------------------------------------------------
__device__ inline void atomicMaxF(float* addr, float val) {
  if (val >= 0.f) atomicMax((int*)addr, __float_as_int(val));
  else            atomicMin((unsigned int*)addr, __float_as_uint(val));
}

__global__ void k_pool(const float4* __restrict__ x4, const float4* __restrict__ h24,
                       const int* __restrict__ gb, float* __restrict__ sm,
                       float* __restrict__ mx) {
  const int QW = (FEAT + HID) / 4;       // 96 quads per node
  const long long total = (long long)NN * QW;
  long long i = (long long)blockIdx.x * blockDim.x + threadIdx.x;
  const long long st = (long long)gridDim.x * blockDim.x;
  for (; i < total; i += st) {
    const long long n = i / QW;
    const int q = (int)(i - n * QW);
    const float4 v = (q < FEAT / 4) ? x4[n * (FEAT / 4) + q]
                                    : h24[n * (HID / 4) + (q - FEAT / 4)];
    const int g = gb[n];
    float* ps = &sm[(long long)g * (FEAT + HID) + q * 4];
    float* pm = &mx[(long long)g * (FEAT + HID) + q * 4];
    atomicAdd(ps + 0, v.x);  atomicMaxF(pm + 0, v.x);
    atomicAdd(ps + 1, v.y);  atomicMaxF(pm + 1, v.y);
    atomicAdd(ps + 2, v.z);  atomicMaxF(pm + 2, v.z);
    atomicAdd(ps + 3, v.w);  atomicMaxF(pm + 3, v.w);
  }
}

__global__ void k_pool_fix(float* mx, long long n) {   // isneginf -> 0
  long long i = (long long)blockIdx.x * blockDim.x + threadIdx.x;
  const long long st = (long long)gridDim.x * blockDim.x;
  for (; i < n; i += st)
    if (mx[i] == -INFINITY) mx[i] = 0.f;
}

// ---------------------------------------------------------------------------
// Stage 8: hid = relu([sm|mx] @ w_lin + b_lin)   G=2048, K=768  (WMMA)
// ---------------------------------------------------------------------------
__global__ void k_lin(const float* __restrict__ sm, const float* __restrict__ mx,
                      const __bf16* __restrict__ wT, const float* __restrict__ blin,
                      float* __restrict__ hid) {
  const int wid = blockIdx.x * (blockDim.x >> 5) + (threadIdx.x >> 5);
  if (wid >= GG / 16) return;
  const int row0 = wid * 16;
  const int SW = FEAT + HID;             // 384
  const int KT = 2 * SW;                 // 768
  v8f c[8];
#pragma unroll
  for (int ot = 0; ot < 8; ++ot) {
    const float bs = blin[ot * 16 + CD_COL];
#pragma unroll
    for (int i = 0; i < 8; ++i) c[ot][i] = bs;
  }
  for (int k0 = 0; k0 < KT; k0 += 32) {
    const float* Abase = (k0 < SW) ? (sm + (long long)row0 * SW + k0)
                                   : (mx + (long long)row0 * SW + (k0 - SW));
    v16bf a = load_a_f32(Abase, SW);
#pragma unroll
    for (int ot = 0; ot < 8; ++ot) {
      v16bf b = load_b_bf16t(wT + (long long)(ot * 16) * KT + k0, KT);
      c[ot] = WMMA_BF16(a, b, c[ot]);
    }
  }
#pragma unroll
  for (int ot = 0; ot < 8; ++ot)
#pragma unroll
    for (int i = 0; i < 8; ++i)
      hid[(long long)(row0 + CD_ROW(i)) * HID + ot * 16 + CD_COL] = fmaxf(c[ot][i], 0.f);
}

// ---------------------------------------------------------------------------
// Stage 9: logits = hid@w_fc + b_fc ; log_softmax — one wave per graph row
// ---------------------------------------------------------------------------
__global__ void k_fc(const float* __restrict__ hid, const float* __restrict__ wfc,
                     const float* __restrict__ bfc, float* __restrict__ out) {
  const int g = blockIdx.x * (blockDim.x >> 5) + (threadIdx.x >> 5);
  const int lane = threadIdx.x & 31;
  if (g >= GG) return;
  float logit = -INFINITY;
  if (lane < CC) {
    float s = bfc[lane];
    for (int k = 0; k < HID; ++k) s += hid[(long long)g * HID + k] * wfc[k * CC + lane];
    logit = s;
  }
  float m = logit;
  for (int off = 16; off; off >>= 1) m = fmaxf(m, __shfl_xor(m, off, 32));
  float e = (lane < CC) ? __expf(logit - m) : 0.f;
  float sum = e;
  for (int off = 16; off; off >>= 1) sum += __shfl_xor(sum, off, 32);
  if (lane < CC) out[(long long)g * CC + lane] = logit - m - __logf(sum);
}

// ---------------------------------------------------------------------------
// Launcher
// ---------------------------------------------------------------------------
extern "C" void kernel_launch(void* const* d_in, const int* in_sizes, int n_in,
                              void* d_out, int out_size, void* d_ws, size_t ws_size,
                              hipStream_t stream) {
  const float* x        = (const float*)d_in[0];
  const float* ew       = (const float*)d_in[1];
  const float* w_rel    = (const float*)d_in[2];
  const float* b_rel    = (const float*)d_in[3];
  const float* w_root   = (const float*)d_in[4];
  const float* comp     = (const float*)d_in[5];
  const float* bases    = (const float*)d_in[6];
  const float* rgcnW    = (const float*)d_in[7];
  const float* rgcnB    = (const float*)d_in[8];
  const float* w_lin    = (const float*)d_in[9];
  const float* b_lin    = (const float*)d_in[10];
  const float* w_fc     = (const float*)d_in[11];
  const float* b_fc     = (const float*)d_in[12];
  const int*   eidx     = (const int*)d_in[13];
  const int*   etype    = (const int*)d_in[14];
  const int*   gbatch   = (const int*)d_in[15];
  const int*   esrc = eidx;
  const int*   etgt = eidx + EE;
  float* out = (float*)d_out;

  // workspace carve-out (256B aligned)
  char* ws = (char*)d_ws;
  size_t off = 0;
  auto carve = [&](size_t bytes) -> char* {
    char* p = ws + off;
    off = (off + bytes + 255) & ~(size_t)255;
    return p;
  };
  float*  agg    = (float*) carve((size_t)NN * FEAT * 4);
  float*  h1     = (float*) carve((size_t)NN * HID * 4);
  float*  xw     = (float*) carve((size_t)NN * RR * HID * 4);
  float*  cnt    = (float*) carve((size_t)NN * RR * 4);
  float*  agg2   = (float*) carve((size_t)NN * HID * 4);
  float*  h2     = (float*) carve((size_t)NN * HID * 4);
  float*  sm     = (float*) carve((size_t)GG * (FEAT + HID) * 4);
  float*  mx     = (float*) carve((size_t)GG * (FEAT + HID) * 4);
  float*  hid    = (float*) carve((size_t)GG * HID * 4);
  __bf16* wrelT  = (__bf16*)carve((size_t)FEAT * HID * 2);
  __bf16* wrootT = (__bf16*)carve((size_t)FEAT * HID * 2);
  __bf16* wrT    = (__bf16*)carve((size_t)RR * HID * HID * 2);
  __bf16* rgT    = (__bf16*)carve((size_t)HID * HID * 2);
  __bf16* wlinT  = (__bf16*)carve((size_t)2 * (FEAT + HID) * HID * 2);

  const int T = 256;

  // zero / -inf init
  k_fill<<<65536, T, 0, stream>>>(agg,  (long long)NN * FEAT, 0.f);
  k_fill<<<4096,  T, 0, stream>>>(cnt,  (long long)NN * RR,   0.f);
  k_fill<<<16384, T, 0, stream>>>(agg2, (long long)NN * HID,  0.f);
  k_fill<<<4096,  T, 0, stream>>>(sm, (long long)GG * (FEAT + HID), 0.f);
  k_fill<<<4096,  T, 0, stream>>>(mx, (long long)GG * (FEAT + HID), -INFINITY);

  // weight conversion f32 -> bf16 (transposed for contiguous B fragments)
  k_cvt_bf16_t<<<(FEAT * HID + T - 1) / T, T, 0, stream>>>(w_rel,  wrelT,  FEAT, HID);
  k_cvt_bf16_t<<<(FEAT * HID + T - 1) / T, T, 0, stream>>>(w_root, wrootT, FEAT, HID);
  k_cvt_bf16_t<<<(HID * HID + T - 1) / T,  T, 0, stream>>>(rgcnW,  rgT,    HID, HID);
  k_cvt_bf16_t<<<(2 * (FEAT + HID) * HID + T - 1) / T, T, 0, stream>>>(
      w_lin, wlinT, 2 * (FEAT + HID), HID);
  k_wr<<<(RR * HID * HID + T - 1) / T, T, 0, stream>>>(comp, bases, wrT);

  // pipeline
  k_edge_agg1<<<65536, T, 0, stream>>>((const float4*)x, ew, esrc, etgt, agg);

  // h1: 3125 waves (one per 16-row strip), 8 waves/block
  k_h1<<<(NN / 16 + 7) / 8, T, 0, stream>>>(agg, x, wrelT, wrootT, b_rel, h1);

  // xw: 3125*16 waves
  k_xw<<<(NN / 16) * RR / 8, T, 0, stream>>>(h1, wrT, xw);

  k_edge_cnt<<<4096, T, 0, stream>>>(etgt, etype, cnt);
  k_edge_agg2<<<65536, T, 0, stream>>>((const float4*)xw, cnt, esrc, etgt, etype, agg2);

  k_h2<<<(NN / 16 + 7) / 8, T, 0, stream>>>(h1, agg2, rgT, rgcnB, h2);

  k_pool<<<32768, T, 0, stream>>>((const float4*)x, (const float4*)h2, gbatch, sm, mx);
  k_pool_fix<<<4096, T, 0, stream>>>(mx, (long long)GG * (FEAT + HID));

  k_lin<<<(GG / 16) / 8, T, 0, stream>>>(sm, mx, wlinT, b_lin, hid);
  k_fc<<<GG / 8, T, 0, stream>>>(hid, w_fc, b_fc, out);
}
